// SeqTrackDecoderXL_84988812853395
// MI455X (gfx1250) — matrix operand order repaired and checked by reading, not compiled, not run on hardware
//
#include <hip/hip_runtime.h>

#define QLEN 1024
#define BSZ 4
#define DM 1024
#define NH 16
#define DH 64
#define TOK (QLEN * BSZ)

typedef _Float16 v8h  __attribute__((ext_vector_type(8)));
typedef _Float16 v16h __attribute__((ext_vector_type(16)));
typedef float    v8f  __attribute__((ext_vector_type(8)));

__device__ __forceinline__ v8f wmma_f16(v16h a, v16h b, v8f c) {
  // D = A(16x32 f16) * B(32x16 f16) + C(16x16 f32)
  return __builtin_amdgcn_wmma_f32_16x16x32_f16(false, a, false, b, (short)0, c,
                                                false, false);
}

__device__ __forceinline__ v16h cat8(v8h lo, v8h hi) {
  return __builtin_shufflevector(lo, hi, 0, 1, 2, 3, 4, 5, 6, 7, 8, 9, 10, 11,
                                 12, 13, 14, 15);
}

// A fragment (16x32, f16), source row-major with leading dim ld.
// Contiguous k order (B uses the same): lane<16 -> M=lane, K=0..15; lane>=16 -> K=16..31
__device__ __forceinline__ v16h load_afrag(const _Float16* A, int ld, int row0,
                                           int k0, int lane) {
  int r  = row0 + (lane & 15);
  int kb = (lane >> 4) << 4;
  const _Float16* p = A + (size_t)r * ld + k0 + kb;
  v8h lo = *(const v8h*)p;
  v8h hi = *(const v8h*)(p + 8);
  return cat8(lo, hi);
}

// B fragment (32x16, f16): operand column n = row n of a row-major [N x K] src.
// Contiguous k order (same as A): lane<16 -> N=lane, K=0..15; lane>=16 -> K=16..31
__device__ __forceinline__ v16h load_bfrag(const _Float16* B, int ld, int col0,
                                           int k0, int lane) {
  const _Float16* p = B + (size_t)(col0 + (lane & 15)) * ld + k0 + ((lane >> 4) << 4);
  v8h lo = *(const v8h*)p;
  v8h hi = *(const v8h*)(p + 8);
  return cat8(lo, hi);
}

// Async 16B-per-lane copy global -> LDS (gfx1250, ASYNCcnt tracked)
__device__ __forceinline__ void async_g2l_b128(unsigned int lds_off,
                                               const _Float16* g) {
  asm volatile("global_load_async_to_lds_b128 %0, %1, off"
               :: "v"(lds_off), "v"(g) : "memory");
}
__device__ __forceinline__ void wait_async0() {
  asm volatile("s_wait_asynccnt 0" ::: "memory");
}

// ---------------------------------------------------------------- converts
__global__ __launch_bounds__(256) void f32_to_f16_kernel(
    const float* __restrict__ src, _Float16* __restrict__ dst, size_t n) {
  size_t i = (size_t)blockIdx.x * blockDim.x + threadIdx.x;
  size_t stride = (size_t)gridDim.x * blockDim.x;
  for (; i < n; i += stride) dst[i] = (_Float16)src[i];
}

// ---------------------------------------------------------------- GEMM
// C[M,N] = A[M,K] * B[N,K]^T ; f32 accumulate; f16 or f32 output.
// Block: 256 thr = 8 waves, wave tile 32x64, block tile 128x128.
// K-step 64, double-buffered LDS tiles filled with async global->LDS copies.
#define KSTEP 64
#define LDP 72  // padded halves per LDS row (144B: conflict-free frag reads)

__global__ __launch_bounds__(256) void gemm_wmma_kernel(
    const _Float16* __restrict__ A, const _Float16* __restrict__ B,
    _Float16* __restrict__ Ch, float* __restrict__ Cf, int M, int N, int K) {
  __shared__ _Float16 sA[2][128 * LDP];
  __shared__ _Float16 sB[2][128 * LDP];
  const int tid  = threadIdx.x;
  const int lane = tid & 31;
  const int wid  = tid >> 5;
  const int wm = wid & 3, wn = wid >> 2;
  const int mblk = blockIdx.x * 128;
  const int nblk = blockIdx.y * 128;
  const int nn = lane & 15, half = lane >> 4;

  // Stage one 128x64 A-tile + 128x64 B-tile: 1024 16B chunks each, 4/thread.
  auto stage = [&](int buf, int k0) {
#pragma unroll
    for (int q = 0; q < 4; q++) {
      int chunk = tid + q * 256;
      int row = chunk >> 3, cir = chunk & 7;  // 8 chunks of 16B per row
      unsigned int la =
          (unsigned int)(uintptr_t)&sA[buf][row * LDP] + cir * 16;
      async_g2l_b128(la, A + (size_t)(mblk + row) * K + k0 + cir * 8);
      unsigned int lb =
          (unsigned int)(uintptr_t)&sB[buf][row * LDP] + cir * 16;
      async_g2l_b128(lb, B + (size_t)(nblk + row) * K + k0 + cir * 8);
    }
  };

  v8f acc[2][4] = {};
  const int nk = K / KSTEP;
  stage(0, 0);
  for (int k = 0; k < nk; k++) {
    wait_async0();
    __syncthreads();
    if (k + 1 < nk) stage((k + 1) & 1, (k + 1) * KSTEP);

    const _Float16* tA = sA[k & 1];
    const _Float16* tB = sB[k & 1];
#pragma unroll
    for (int kk = 0; kk < 2; kk++) {
      const int kloc = kk * 32;
      v16h a0 = load_afrag(tA, LDP, wm * 32, kloc, lane);
      v16h a1 = load_afrag(tA, LDP, wm * 32 + 16, kloc, lane);
      v16h b0 = load_bfrag(tB, LDP, wn * 64, kloc, lane);
      v16h b1 = load_bfrag(tB, LDP, wn * 64 + 16, kloc, lane);
      v16h b2 = load_bfrag(tB, LDP, wn * 64 + 32, kloc, lane);
      v16h b3 = load_bfrag(tB, LDP, wn * 64 + 48, kloc, lane);
      acc[0][0] = wmma_f16(a0, b0, acc[0][0]);
      acc[0][1] = wmma_f16(a0, b1, acc[0][1]);
      acc[0][2] = wmma_f16(a0, b2, acc[0][2]);
      acc[0][3] = wmma_f16(a0, b3, acc[0][3]);
      acc[1][0] = wmma_f16(a1, b0, acc[1][0]);
      acc[1][1] = wmma_f16(a1, b1, acc[1][1]);
      acc[1][2] = wmma_f16(a1, b2, acc[1][2]);
      acc[1][3] = wmma_f16(a1, b3, acc[1][3]);
    }
  }

  const int m0 = mblk + wm * 32;
  const int n0 = nblk + wn * 64;
#pragma unroll
  for (int i = 0; i < 2; i++)
#pragma unroll
    for (int j = 0; j < 4; j++)
#pragma unroll
      for (int g = 0; g < 8; g++) {
        int m = m0 + i * 16 + g + (half << 3);
        int n = n0 + j * 16 + nn;
        float v = acc[i][j][g];
        if (Cf) Cf[(size_t)m * N + n] = v;
        else    Ch[(size_t)m * N + n] = (_Float16)v;
      }
}

// ---------------------------------------------------------------- pack
__global__ __launch_bounds__(256) void pack_kernel(
    const _Float16* __restrict__ qkv, const _Float16* __restrict__ rkh,
    const float* __restrict__ rwb, const float* __restrict__ rrb,
    _Float16* __restrict__ qa, _Float16* __restrict__ qb,
    _Float16* __restrict__ kp, _Float16* __restrict__ rkp,
    _Float16* __restrict__ vT) {
  const int bh = blockIdx.x;
  const int b = bh >> 4, h = bh & 15;
  const int d  = threadIdx.x & 63;
  const int jj = threadIdx.x >> 6;
  const int j  = blockIdx.y * 4 + jj;
  const size_t tok = (size_t)j * BSZ + b;
  const float scale = 0.125f;  // 1/sqrt(64)

  float qv = (float)qkv[tok * (3 * DM) + h * DH + d];
  size_t o = ((size_t)bh * QLEN + j) * DH + d;
  qa[o]  = (_Float16)((qv + rwb[h * DH + d]) * scale);
  qb[o]  = (_Float16)((qv + rrb[h * DH + d]) * scale);
  kp[o]  = qkv[tok * (3 * DM) + DM + h * DH + d];
  rkp[o] = rkh[tok * DM + h * DH + d];
  vT[((size_t)bh * DH + d) * QLEN + j] = qkv[tok * (3 * DM) + 2 * DM + h * DH + d];
}

// ---------------------------------------------------------------- attention
// One block = one (batch,head) x 128 queries; 8 waves x 16 queries each.
// Flash-style online softmax; rel-shift handled via a 16x48 WMMA "band".
__global__ __launch_bounds__(256) void attn_kernel(
    const _Float16* __restrict__ qa, const _Float16* __restrict__ qb,
    const _Float16* __restrict__ kp, const _Float16* __restrict__ rkp,
    const _Float16* __restrict__ vT, _Float16* __restrict__ avec) {
  __shared__ float    s_band[8][16 * 48];
  __shared__ _Float16 s_p[8][16 * 32];
  const int lane = threadIdx.x & 31;
  const int wid  = threadIdx.x >> 5;
  const int nn = lane & 15, half = lane >> 4;
  const int bh = blockIdx.x;
  const int b = bh >> 4, h = bh & 15;
  const int i0 = blockIdx.y * 128 + wid * 16;

  const _Float16* QA = qa  + (size_t)bh * QLEN * DH;
  const _Float16* QB = qb  + (size_t)bh * QLEN * DH;
  const _Float16* KP = kp  + (size_t)bh * QLEN * DH;
  const _Float16* RK = rkp + (size_t)bh * QLEN * DH;
  const _Float16* VT = vT  + (size_t)bh * DH * QLEN;

  v16h aQ[2], aB[2];
#pragma unroll
  for (int kk = 0; kk < 2; kk++) {
    aQ[kk] = load_afrag(QA, DH, i0, kk * 32, lane);
    aB[kk] = load_afrag(QB, DH, i0, kk * 32, lane);
  }

  v8f O[4] = {};
  float rmax[8], rsum[8];
#pragma unroll
  for (int g = 0; g < 8; g++) { rmax[g] = -1e30f; rsum[g] = 0.f; }

  for (int j0 = 0; j0 <= i0 + 15; j0 += 32) {
    // AC term: (q+rwb)*scale . K^T
    v8f S[2] = {};
#pragma unroll
    for (int kk = 0; kk < 2; kk++)
#pragma unroll
      for (int t = 0; t < 2; t++)
        S[t] = wmma_f16(aQ[kk], load_bfrag(KP, DH, j0 + t * 16, kk * 32, lane), S[t]);

    // BD band: band[m][c] = qb[i0+m] . rk[base+c], rk row = QLEN-1-(i0+m)+j
    v8f Bd[3] = {};
    const int base = (QLEN - 16) - i0 + j0;
#pragma unroll
    for (int kk = 0; kk < 2; kk++)
#pragma unroll
      for (int t = 0; t < 3; t++) {
        int rr = base + t * 16 + nn;
        rr = rr < 0 ? 0 : (rr > QLEN - 1 ? QLEN - 1 : rr);  // clamped rows masked
        const _Float16* p = RK + (size_t)rr * DH + kk * 32 + (half << 4);
        v8h lo = *(const v8h*)p;
        v8h hi = *(const v8h*)(p + 8);
        Bd[t] = wmma_f16(aB[kk], cat8(lo, hi), Bd[t]);
      }
#pragma unroll
    for (int t = 0; t < 3; t++)
#pragma unroll
      for (int g = 0; g < 8; g++)
        s_band[wid][(g + half * 8) * 48 + t * 16 + nn] = Bd[t][g];
    asm volatile("s_wait_dscnt 0" ::: "memory");

    // fuse band (rel-shift) + causal mask
    float sv[2][8];
#pragma unroll
    for (int t = 0; t < 2; t++)
#pragma unroll
      for (int g = 0; g < 8; g++) {
        int m = g + half * 8;
        int c = t * 16 + nn + 15 - m;  // (j-j0) + 15 - m
        float v = S[t][g] + s_band[wid][m * 48 + c];
        int j = j0 + t * 16 + nn;
        if (j > i0 + m) v = -1e30f;
        sv[t][g] = v;
      }

    // online softmax (rows live in one 16-lane half; reduce with xor<16)
#pragma unroll
    for (int g = 0; g < 8; g++) {
      float v = fmaxf(sv[0][g], sv[1][g]);
#pragma unroll
      for (int off = 1; off < 16; off <<= 1) v = fmaxf(v, __shfl_xor(v, off, 32));
      float nm = fmaxf(rmax[g], v);
      float sc = __expf(rmax[g] - nm);
      rmax[g] = nm;
      float p0 = __expf(sv[0][g] - nm);
      float p1 = __expf(sv[1][g] - nm);
      float ls = p0 + p1;
#pragma unroll
      for (int off = 1; off < 16; off <<= 1) ls += __shfl_xor(ls, off, 32);
      rsum[g] = rsum[g] * sc + ls;
#pragma unroll
      for (int dt = 0; dt < 4; dt++) O[dt][g] *= sc;
      sv[0][g] = p0; sv[1][g] = p1;
    }

    // restage P (C-layout) -> A-fragment layout via wave-private LDS
#pragma unroll
    for (int t = 0; t < 2; t++)
#pragma unroll
      for (int g = 0; g < 8; g++)
        s_p[wid][(g + half * 8) * 32 + t * 16 + nn] = (_Float16)sv[t][g];
    asm volatile("s_wait_dscnt 0" ::: "memory");

    {
      int kb = half << 4;
      const _Float16* pp = &s_p[wid][nn * 32 + kb];
      v8h lo = *(const v8h*)pp;
      v8h hi = *(const v8h*)(pp + 8);
      v16h pf = cat8(lo, hi);
#pragma unroll
      for (int dt = 0; dt < 4; dt++) {
        const _Float16* pv = VT + (size_t)(dt * 16 + nn) * QLEN + j0 + (half << 4);
        v8h vlo = *(const v8h*)pv;
        v8h vhi = *(const v8h*)(pv + 8);
        O[dt] = wmma_f16(pf, cat8(vlo, vhi), O[dt]);
      }
    }
  }

#pragma unroll
  for (int g = 0; g < 8; g++) {
    float inv = 1.f / rsum[g];
    int i = i0 + g + half * 8;
#pragma unroll
    for (int dt = 0; dt < 4; dt++) {
      int d = dt * 16 + nn;
      avec[((size_t)i * BSZ + b) * DM + h * DH + d] = (_Float16)(O[dt][g] * inv);
    }
  }
}

// ---------------------------------------------------------------- layernorm
__global__ __launch_bounds__(256) void ln_kernel(
    const float* __restrict__ w, const float* __restrict__ aout,
    const float* __restrict__ gamma, const float* __restrict__ beta,
    float* __restrict__ out) {
  __shared__ float ws1[8], ws2[8];
  const int row = blockIdx.x;
  const int tid = threadIdx.x;
  const int lane = tid & 31, wid = tid >> 5;
  float xv[4], s = 0.f, s2 = 0.f;
#pragma unroll
  for (int k = 0; k < 4; k++) {
    int c = k * 256 + tid;
    float x = w[(size_t)row * DM + c] + aout[(size_t)row * DM + c];
    xv[k] = x; s += x; s2 += x * x;
  }
#pragma unroll
  for (int off = 1; off < 32; off <<= 1) {
    s += __shfl_xor(s, off, 32);
    s2 += __shfl_xor(s2, off, 32);
  }
  if (lane == 0) { ws1[wid] = s; ws2[wid] = s2; }
  __syncthreads();
  float ts = 0.f, ts2 = 0.f;
#pragma unroll
  for (int i = 0; i < 8; i++) { ts += ws1[i]; ts2 += ws2[i]; }
  float mu  = ts * (1.f / DM);
  float var = ts2 * (1.f / DM) - mu * mu;
  float inv = rsqrtf(var + 1e-5f);
#pragma unroll
  for (int k = 0; k < 4; k++) {
    int c = k * 256 + tid;
    out[(size_t)row * DM + c] = (xv[k] - mu) * inv * gamma[c] + beta[c];
  }
}

// ---------------------------------------------------------------- launch
extern "C" void kernel_launch(void* const* d_in, const int* in_sizes, int n_in,
                              void* d_out, int out_size, void* d_ws,
                              size_t ws_size, hipStream_t stream) {
  const float* w     = (const float*)d_in[0];
  const float* r     = (const float*)d_in[1];
  const float* rwb   = (const float*)d_in[2];
  const float* rrb   = (const float*)d_in[3];
  const float* qkvw  = (const float*)d_in[4];
  const float* rnetw = (const float*)d_in[5];
  const float* ow    = (const float*)d_in[6];
  const float* gamma = (const float*)d_in[7];
  const float* beta  = (const float*)d_in[8];
  float* out = (float*)d_out;

  char* ws = (char*)d_ws;
  size_t off = 0;
  auto alloc = [&](size_t bytes) -> char* {
    char* p = ws + off;
    off += (bytes + 255) & ~(size_t)255;
    return p;
  };
  _Float16* h_w     = (_Float16*)alloc((size_t)TOK * DM * 2);
  _Float16* h_r     = (_Float16*)alloc((size_t)TOK * DM * 2);
  _Float16* h_qkvw  = (_Float16*)alloc((size_t)3 * DM * DM * 2);
  _Float16* h_rnetw = (_Float16*)alloc((size_t)DM * DM * 2);
  _Float16* h_ow    = (_Float16*)alloc((size_t)DM * DM * 2);
  _Float16* qkv_h   = (_Float16*)alloc((size_t)TOK * 3 * DM * 2);
  _Float16* rk_h    = (_Float16*)alloc((size_t)TOK * DM * 2);
  _Float16* qa      = (_Float16*)alloc((size_t)BSZ * NH * QLEN * DH * 2);
  _Float16* qb      = (_Float16*)alloc((size_t)BSZ * NH * QLEN * DH * 2);
  _Float16* kp      = (_Float16*)alloc((size_t)BSZ * NH * QLEN * DH * 2);
  _Float16* rkp     = (_Float16*)alloc((size_t)BSZ * NH * QLEN * DH * 2);
  _Float16* vT      = (_Float16*)alloc((size_t)BSZ * NH * DH * QLEN * 2);
  _Float16* avec    = (_Float16*)alloc((size_t)TOK * DM * 2);
  float*    aout    = (float*)alloc((size_t)TOK * DM * 4);
  (void)ws_size; (void)in_sizes; (void)n_in; (void)out_size;

  // 1) f32 -> f16
  f32_to_f16_kernel<<<1024, 256, 0, stream>>>(w, h_w, (size_t)TOK * DM);
  f32_to_f16_kernel<<<1024, 256, 0, stream>>>(r, h_r, (size_t)TOK * DM);
  f32_to_f16_kernel<<<1024, 256, 0, stream>>>(qkvw, h_qkvw, (size_t)3 * DM * DM);
  f32_to_f16_kernel<<<1024, 256, 0, stream>>>(rnetw, h_rnetw, (size_t)DM * DM);
  f32_to_f16_kernel<<<1024, 256, 0, stream>>>(ow, h_ow, (size_t)DM * DM);

  // 2) qkv = W @ qkv_w^T   [4096 x 3072]
  gemm_wmma_kernel<<<dim3(TOK / 128, (3 * DM) / 128), 256, 0, stream>>>(
      h_w, h_qkvw, qkv_h, nullptr, TOK, 3 * DM, DM);
  // 3) r_head_k = R @ r_net_w^T   [4096 x 1024]
  gemm_wmma_kernel<<<dim3(TOK / 128, DM / 128), 256, 0, stream>>>(
      h_r, h_rnetw, rk_h, nullptr, TOK, DM, DM);

  // 4) per-(b,head) packing: bias+scale fold, V transpose
  pack_kernel<<<dim3(BSZ * NH, QLEN / 4), 256, 0, stream>>>(
      qkv_h, rk_h, rwb, rrb, qa, qb, kp, rkp, vT);

  // 5) rel-attention, flash-style
  attn_kernel<<<dim3(BSZ * NH, QLEN / 128), 256, 0, stream>>>(qa, qb, kp, rkp,
                                                              vT, avec);

  // 6) attn_out = attn_vec @ o_w^T  (f32 out)
  gemm_wmma_kernel<<<dim3(TOK / 128, DM / 128), 256, 0, stream>>>(
      avec, h_ow, nullptr, aout, TOK, DM, DM);

  // 7) residual + layernorm
  ln_kernel<<<dim3(TOK), 256, 0, stream>>>(w, aout, gamma, beta, out);
}
